// GATBaseline_77129022702013
// MI455X (gfx1250) — matrix operand order repaired and checked by reading, not compiled, not run on hardware
//
#include <hip/hip_runtime.h>
#include <hip/hip_bf16.h>

// ---- problem constants (from reference) ----
#define NN   50000
#define EE   800000
#define ENE  850000      // E + N self loops
#define HC   256         // H*C
#define CC   64
#define HH   4
#define GG   4

typedef __attribute__((ext_vector_type(16))) __bf16 v16bf;
typedef __attribute__((ext_vector_type(8)))  float  v8f;

union FragBF {
    uint4  q[2];   // 32 bytes
    v16bf  v;
};

__device__ __forceinline__ unsigned short f2bf(float f) {
    unsigned u = __float_as_uint(f);
    unsigned r = (u + 0x7FFFu + ((u >> 16) & 1u)) >> 16;   // round-to-nearest-even
    return (unsigned short)r;
}
// monotone float <-> uint key for atomicMax over floats (handles negatives)
__device__ __forceinline__ unsigned fkey(float f) {
    unsigned u = __float_as_uint(f);
    return (u & 0x80000000u) ? ~u : (u | 0x80000000u);
}
__device__ __forceinline__ float funkey(unsigned k) {
    unsigned u = (k & 0x80000000u) ? (k & 0x7FFFFFFFu) : ~k;
    return __uint_as_float(u);
}

// ---------------- pack kernels (fp32 -> bf16) ----------------
__global__ void pack_act_k(const float* __restrict__ A, unsigned short* __restrict__ O, long tot) {
    long i = (long)blockIdx.x * blockDim.x + threadIdx.x;
    if (i < tot) O[i] = f2bf(A[i]);
}
// W [K,M] row-major -> WT [M,K] bf16 (column-major of W)
__global__ void pack_wT_k(const float* __restrict__ W, unsigned short* __restrict__ WT, int K, int M) {
    int i = blockIdx.x * blockDim.x + threadIdx.x;
    if (i >= K * M) return;
    int k = i / M, m = i - k * M;
    WT[(long)m * K + k] = f2bf(W[i]);
}

// ---------------- WMMA GEMM: C[N,M] = A[N,K] * W[K,M]  (A bf16 rowmajor, BT = W^T bf16) ----------------
__global__ void wmma_gemm_k(const unsigned short* __restrict__ A,
                            const unsigned short* __restrict__ BT,
                            float* __restrict__ C,
                            int Nrows, int K, int Mcols) {
    const int lane = threadIdx.x & 31;
    const int wave = threadIdx.x >> 5;
    const int ntc  = Mcols >> 4;                       // col tiles
    long tile = (long)blockIdx.x * (blockDim.x >> 5) + wave;
    long total = (long)(Nrows >> 4) * ntc;             // Nrows % 16 == 0 here
    if (tile >= total) return;
    const int mt = (int)(tile / ntc);
    const int nt = (int)(tile - (long)mt * ntc);

    const int r    = lane & 15;
    const int half = lane >> 4;
    const unsigned short* arow = A  + (long)(mt * 16 + r) * K;
    const unsigned short* bcol = BT + (long)(nt * 16 + r) * K;

    v8f acc = {};
    for (int k0 = 0; k0 < K; k0 += 32) {
        FragBF fa, fb;
        // A 16x32 bf16 fragment: lanes<16 hold K 0..7 & 16..23 of k-block, lanes>=16 hold 8..15 & 24..31
        fa.q[0] = *(const uint4*)(arow + k0 + half * 8);
        fa.q[1] = *(const uint4*)(arow + k0 + 16 + half * 8);
        // B 32x16 bf16 fragment: col = lane&15, lanes<16 hold K 0..7 & 16..23, lanes>=16 hold 8..15 & 24..31
        fb.q[0] = *(const uint4*)(bcol + k0 + half * 8);
        fb.q[1] = *(const uint4*)(bcol + k0 + 16 + half * 8);
        acc = __builtin_amdgcn_wmma_f32_16x16x32_bf16(false, fa.v, false, fb.v,
                                                      (short)0, acc, false, false);
    }
    // D: VGPR i -> row (half*8 + i), col = lane&15
    float* crow = C + (long)(mt * 16 + half * 8) * Mcols + nt * 16 + r;
#pragma unroll
    for (int i = 0; i < 8; ++i) crow[(long)i * Mcols] = acc[i];
}

// ---------------- attention logits: al[n,h] = dot(xp[n,h,:], a[h,:]) ----------------
__global__ void logits_k(const float* __restrict__ xp, const float* __restrict__ asrc,
                         const float* __restrict__ adst,
                         float* __restrict__ als, float* __restrict__ ald, int n) {
    int i = blockIdx.x * blockDim.x + threadIdx.x;     // i = node*H + h
    if (i >= n * HH) return;
    int h = i & 3;
    const float* v  = xp + (long)(i >> 2) * HC + h * CC;
    const float* as_ = asrc + h * CC;
    const float* ad_ = adst + h * CC;
    float s1 = 0.f, s2 = 0.f;
#pragma unroll 8
    for (int c = 0; c < CC; ++c) { float x = v[c]; s1 += x * as_[c]; s2 += x * ad_[c]; }
    als[i] = s1; ald[i] = s2;
}

// ---------------- per-layer init: m=-inf, s=0, agg=0 ----------------
__global__ void init_layer_k(unsigned* __restrict__ m_u, float* __restrict__ sbuf,
                             float* __restrict__ agg) {
    long i = (long)blockIdx.x * 256 + threadIdx.x;
    if (i < (long)NN * HH) { m_u[i] = fkey(-3.0e38f); sbuf[i] = 0.f; }
    if (i < (long)NN * HC) agg[i] = 0.f;
}

// ---------------- edge pass 1: e = leakyrelu(als[src]+ald[dst]); seg-max ----------------
__global__ void edge_max_k(const int* __restrict__ ei,
                           const float* __restrict__ als, const float* __restrict__ ald,
                           float* __restrict__ eW, unsigned* __restrict__ m_u) {
    int eid = blockIdx.x * blockDim.x + threadIdx.x;
    if (eid >= ENE) return;
    int s = (eid < EE) ? ei[eid]       : (eid - EE);
    int d = (eid < EE) ? ei[EE + eid]  : (eid - EE);
#pragma unroll
    for (int h = 0; h < HH; ++h) {
        float e = als[s * HH + h] + ald[d * HH + h];
        e = e > 0.f ? e : 0.2f * e;                    // leaky_relu 0.2
        eW[(long)eid * HH + h] = e;
        atomicMax(&m_u[d * HH + h], fkey(e));
    }
}

// ---------------- edge pass 2: ex = exp(e - m[dst]); seg-sum ----------------
__global__ void edge_expsum_k(const int* __restrict__ ei,
                              float* __restrict__ eW, const unsigned* __restrict__ m_u,
                              float* __restrict__ sbuf) {
    int eid = blockIdx.x * blockDim.x + threadIdx.x;
    if (eid >= ENE) return;
    int d = (eid < EE) ? ei[EE + eid] : (eid - EE);
#pragma unroll
    for (int h = 0; h < HH; ++h) {
        float mx = funkey(m_u[d * HH + h]);
        float ex = __expf(eW[(long)eid * HH + h] - mx);
        eW[(long)eid * HH + h] = ex;
        atomicAdd(&sbuf[d * HH + h], ex);
    }
}

// ---------------- edge pass 3: agg[dst] += alpha * xp[src]  (one 256-thread block / edge) ----------------
__global__ void edge_agg_k(const int* __restrict__ ei,
                           const float* __restrict__ eW, const float* __restrict__ sbuf,
                           const float* __restrict__ xp, float* __restrict__ agg) {
    int eid = blockIdx.x;
    int t = threadIdx.x;                               // 0..255, h = t>>6
    int s = (eid < EE) ? ei[eid]      : (eid - EE);
    int d = (eid < EE) ? ei[EE + eid] : (eid - EE);
    int h = t >> 6;
    float alpha = eW[(long)eid * HH + h] / (sbuf[d * HH + h] + 1e-16f);
    atomicAdd(&agg[(long)d * HC + t], alpha * xp[(long)s * HC + t]);
}

// ---------------- post: h = elu(agg + b)  (layers 1,2) ----------------
__global__ void post_elu_k(const float* __restrict__ agg, const float* __restrict__ b,
                           float* __restrict__ hout, long tot) {
    long i = (long)blockIdx.x * 256 + threadIdx.x;
    if (i >= tot) return;
    float v = agg[i] + b[i & (HC - 1)];
    hout[i] = v > 0.f ? v : (__expf(v) - 1.f);
}

// ---------------- post: h3 = mean_heads(agg) + b3 (layer 3) ----------------
__global__ void post_mean_k(const float* __restrict__ agg, const float* __restrict__ b3,
                            float* __restrict__ h3) {
    int n = blockIdx.x, c = threadIdx.x;               // blockDim = 64
    float v = 0.f;
#pragma unroll
    for (int h = 0; h < HH; ++h) v += agg[(long)n * HC + h * CC + c];
    h3[(long)n * CC + c] = v * 0.25f + b3[c];
}

// ---------------- pooling ----------------
__global__ void pool_init_k(float* psum, unsigned* pmax, float* cnt) {
    int i = threadIdx.x;
    if (i < GG * CC) { psum[i] = 0.f; pmax[i] = fkey(-3.0e38f); }
    if (i < GG) cnt[i] = 0.f;
}
__global__ void pool_k(const float* __restrict__ h3, const int* __restrict__ batch,
                       float* psum, unsigned* pmax, float* cnt) {
    int n = blockIdx.x, c = threadIdx.x;               // blockDim = 64
    int g = batch[n];
    float v = h3[(long)n * CC + c];
    atomicAdd(&psum[g * CC + c], v);
    atomicMax(&pmax[g * CC + c], fkey(v));
    if (c == 0) atomicAdd(&cnt[g], 1.0f);
}

// ---------------- tiny MLP on 4 graphs (single block) ----------------
__global__ void mlp_k(const float* __restrict__ psum, const unsigned* __restrict__ pmax,
                      const float* __restrict__ cnt,
                      const float* __restrict__ w1, const float* __restrict__ b1,
                      const float* __restrict__ w2, const float* __restrict__ b2,
                      const float* __restrict__ w3, const float* __restrict__ b3,
                      float* __restrict__ out) {
    __shared__ float gf[GG][2 * CC];
    __shared__ float l1[GG][64];
    __shared__ float l2[GG][32];
    int t = threadIdx.x;                                // 128 threads
    for (int g = 0; g < GG; ++g) {
        if (t < CC) {
            float c  = cnt[g];
            float mn = psum[g * CC + t] / fmaxf(c, 1.0f);
            float mx = funkey(pmax[g * CC + t]);
            if (!(c > 0.f)) mx = 0.f;
            gf[g][t] = mn; gf[g][CC + t] = mx;
        }
    }
    __syncthreads();
    for (int g = 0; g < GG; ++g)
        if (t < 64) {
            float s = b1[t];
            for (int k = 0; k < 2 * CC; ++k) s += gf[g][k] * w1[k * 64 + t];
            l1[g][t] = fmaxf(s, 0.f);
        }
    __syncthreads();
    for (int g = 0; g < GG; ++g)
        if (t < 32) {
            float s = b2[t];
            for (int k = 0; k < 64; ++k) s += l1[g][k] * w2[k * 32 + t];
            l2[g][t] = fmaxf(s, 0.f);
        }
    __syncthreads();
    if (t < GG * 2) {
        int g = t >> 1, o = t & 1;
        float s = b3[o];
        for (int k = 0; k < 32; ++k) s += l2[g][k] * w3[k * 2 + o];
        out[g * 2 + o] = s;
    }
}

extern "C" void kernel_launch(void* const* d_in, const int* in_sizes, int n_in,
                              void* d_out, int out_size, void* d_ws, size_t ws_size,
                              hipStream_t stream) {
    const float* x     = (const float*)d_in[0];
    const int*   ei    = (const int*)d_in[1];
    const int*   batch = (const int*)d_in[2];
    const float* W1 = (const float*)d_in[3];
    const float* as1 = (const float*)d_in[4]; const float* ad1 = (const float*)d_in[5];
    const float* b1  = (const float*)d_in[6];
    const float* W2 = (const float*)d_in[7];
    const float* as2 = (const float*)d_in[8]; const float* ad2 = (const float*)d_in[9];
    const float* b2  = (const float*)d_in[10];
    const float* W3 = (const float*)d_in[11];
    const float* as3 = (const float*)d_in[12]; const float* ad3 = (const float*)d_in[13];
    const float* b3  = (const float*)d_in[14];
    const float* wm1 = (const float*)d_in[15]; const float* bm1 = (const float*)d_in[16];
    const float* wm2 = (const float*)d_in[17]; const float* bm2 = (const float*)d_in[18];
    const float* wm3 = (const float*)d_in[19]; const float* bm3 = (const float*)d_in[20];
    float* out = (float*)d_out;

    // carve workspace
    char* w = (char*)d_ws;
    auto carve = [&](size_t bytes) -> void* {
        void* p = (void*)w; w += (bytes + 255) & ~(size_t)255; return p;
    };
    float*          xp   = (float*)carve((size_t)NN * HC * 4);
    float*          agg  = (float*)carve((size_t)NN * HC * 4);
    float*          hbuf = (float*)carve((size_t)NN * HC * 4);
    float*          h3   = (float*)carve((size_t)NN * CC * 4);
    float*          als  = (float*)carve((size_t)NN * HH * 4);
    float*          ald  = (float*)carve((size_t)NN * HH * 4);
    unsigned*       m_u  = (unsigned*)carve((size_t)NN * HH * 4);
    float*          sbuf = (float*)carve((size_t)NN * HH * 4);
    float*          eW   = (float*)carve((size_t)ENE * HH * 4);
    float*          psum = (float*)carve((size_t)GG * CC * 4);
    unsigned*       pmax = (unsigned*)carve((size_t)GG * CC * 4);
    float*          cnt  = (float*)carve((size_t)GG * 4);
    unsigned short* abf  = (unsigned short*)carve((size_t)NN * HC * 2);
    unsigned short* wtb  = (unsigned short*)carve((size_t)HC * HC * 2);

    auto run_layer = [&](const float* actin, int K, const float* W,
                         const float* asv, const float* adv, const float* bias, int layer) {
        long totA = (long)NN * K;
        pack_act_k<<<(unsigned)((totA + 255) / 256), 256, 0, stream>>>(actin, abf, totA);
        pack_wT_k<<<(K * HC + 255) / 256, 256, 0, stream>>>(W, wtb, K, HC);
        // 3125 row-tiles x 16 col-tiles = 50000 wave-tiles, 8 waves/block
        wmma_gemm_k<<<6250, 256, 0, stream>>>(abf, wtb, xp, NN, K, HC);
        logits_k<<<(NN * HH + 255) / 256, 256, 0, stream>>>(xp, asv, adv, als, ald, NN);
        init_layer_k<<<NN, 256, 0, stream>>>(m_u, sbuf, agg);
        edge_max_k<<<(ENE + 255) / 256, 256, 0, stream>>>(ei, als, ald, eW, m_u);
        edge_expsum_k<<<(ENE + 255) / 256, 256, 0, stream>>>(ei, eW, m_u, sbuf);
        edge_agg_k<<<ENE, 256, 0, stream>>>(ei, eW, sbuf, xp, agg);
        if (layer < 3)
            post_elu_k<<<NN, 256, 0, stream>>>(agg, bias, hbuf, (long)NN * HC);
        else
            post_mean_k<<<NN, 64, 0, stream>>>(agg, bias, h3);
    };

    run_layer(x,    128, W1, as1, ad1, b1, 1);
    run_layer(hbuf, 256, W2, as2, ad2, b2, 2);
    run_layer(hbuf, 256, W3, as3, ad3, b3, 3);

    pool_init_k<<<1, 256, 0, stream>>>(psum, pmax, cnt);
    pool_k<<<NN, 64, 0, stream>>>(h3, batch, psum, pmax, cnt);
    mlp_k<<<1, 128, 0, stream>>>(psum, pmax, cnt, wm1, bm1, wm2, bm2, wm3, bm3, out);
}
